// OGTree_13099650253026
// MI455X (gfx1250) — matrix operand order repaired and checked by reading, not compiled, not run on hardware
//
#include <hip/hip_runtime.h>
#include <stdint.h>

#define BATCH      131072
#define NFEAT      256
#define NOUT       64
#define N_INTERNAL 1023
#define TILE_S     256          // samples per workgroup
#define XS_STRIDE  264          // ushort stride (K padded: 528B rows, 16B aligned)
#define SC_STRIDE  65           // padded f32 score row (conflict-free column reads)

// ---- LDS layout (dynamic shared) ----
#define OFF_XS   0                      // 256*264*2 = 135168
#define OFF_WST  135168                 // 64*264*2  =  33792
#define OFF_SC   168960                 // 256*65*4  =  66560
#define OFF_MLT  235520                 // 256*4
#define OFF_LFI  236544                 // 256*4
#define SMEM_BYTES 237568

typedef __bf16 bf16_t;
typedef bf16_t v16bf __attribute__((ext_vector_type(16)));
typedef bf16_t v8bf  __attribute__((ext_vector_type(8)));
typedef float  v8f   __attribute__((ext_vector_type(8)));
typedef int    v4i_g __attribute__((vector_size(16)));   // matches builtin's V4i pointee

__device__ __forceinline__ unsigned short f2bf(float f) {
  unsigned u = __float_as_uint(f);
  u += 0x7fffu + ((u >> 16) & 1u);          // round-to-nearest-even
  return (unsigned short)(u >> 16);
}
__device__ __forceinline__ float bf_lo(unsigned u){ return __uint_as_float(u << 16); }
__device__ __forceinline__ float bf_hi(unsigned u){ return __uint_as_float(u & 0xffff0000u); }

// ---- async global->LDS 16B copy (CDNA5 GLOBAL_LOAD_ASYNC_TO_LDS_B128, ASYNCcnt) ----
__device__ __forceinline__ void async_g2l_b128(const void* gsrc, void* ldst) {
#if __has_builtin(__builtin_amdgcn_global_load_async_to_lds_b128)
  __builtin_amdgcn_global_load_async_to_lds_b128(
      (__attribute__((address_space(1))) v4i_g*)(unsigned long long)(size_t)gsrc,
      (__attribute__((address_space(3))) v4i_g*)(unsigned)(size_t)ldst,  // low32 of generic shared ptr == LDS offset
      0, 0);
#else
  unsigned loff = (unsigned)(size_t)ldst;
  unsigned long long ga = (unsigned long long)(size_t)gsrc;
  asm volatile("global_load_async_to_lds_b128 %0, %1, off"
               :: "v"(loff), "v"(ga) : "memory");
#endif
}
__device__ __forceinline__ void wait_async0() {
#if __has_builtin(__builtin_amdgcn_s_wait_asynccnt)
  __builtin_amdgcn_s_wait_asynccnt(0);
#else
  asm volatile("s_wait_asynccnt 0" ::: "memory");
#endif
}

// ---- pre-pass: W f32 -> bf16 (makes both WMMA B-operand and halves deep-level gather bytes) ----
__global__ __launch_bounds__(256) void wconv_kernel(const float* __restrict__ W,
                                                    unsigned short* __restrict__ wbf, int n) {
  int i = blockIdx.x * 256 + threadIdx.x;
  if (i < n) wbf[i] = f2bf(W[i]);
}

__global__ __launch_bounds__(256) void ogtree_kernel(
    const float* __restrict__ x, const unsigned short* __restrict__ wbf,
    const float* __restrict__ bias, const float* __restrict__ alphav,
    const float* __restrict__ leaves, float* __restrict__ out)
{
  extern __shared__ char smem[];
  unsigned short* xs  = (unsigned short*)(smem + OFF_XS);   // [256][264] bf16 x tile
  unsigned short* wst = (unsigned short*)(smem + OFF_WST);  // [64][264]  bf16 W rows 0..63
  float*          sc  = (float*)(smem + OFF_SC);            // [256][65]  dense scores
  float*          mlt = (float*)(smem + OFF_MLT);
  int*            lfi = (int*)(smem + OFF_LFI);

  const int tid  = threadIdx.x;
  const int lane = tid & 31;
  const int wave = tid >> 5;
  const size_t tile_base = (size_t)blockIdx.x * TILE_S;

  // ---- stage W[0:64) into LDS via async global->LDS path (2048 x 16B chunks) ----
  #pragma unroll
  for (int i = 0; i < 8; ++i) {
    int c   = tid + i * 256;
    int row = c >> 5;
    int col = (c & 31) * 8;                       // ushort units, 16B chunks
    async_g2l_b128(wbf + row * NFEAT + col, wst + row * XS_STRIDE + col);
  }

  // ---- stage x tile: coalesced float4 reads -> bf16 LDS (x read once from HBM) ----
  const float4* xg4 = (const float4*)(x + tile_base * NFEAT);
  #pragma unroll 4
  for (int i = 0; i < 64; ++i) {
    int v = tid + i * 256;
    float4 f = xg4[v];
    int s = v >> 6;
    int k = (v & 63) * 4;
    unsigned p0 = (unsigned)f2bf(f.x) | ((unsigned)f2bf(f.y) << 16);
    unsigned p1 = (unsigned)f2bf(f.z) | ((unsigned)f2bf(f.w) << 16);
    *(uint2*)(xs + s * XS_STRIDE + k) = make_uint2(p0, p1);
  }

  wait_async0();
  __syncthreads();

  // ---- dense phase: S[256x64] = Xtile * W[0:64)^T via v_wmma_f32_16x16x32_bf16 ----
  // wave handles M-tiles {wave*32, wave*32+16}; 4 N-groups of 16 nodes; K in 8 steps of 32.
  // Software-pipelined: fragments for k+1 are issued before consuming k's, so the
  // scheduler can overlap ds_loads with the WMMA chain instead of s_wait_dscnt 0.
  const int hi = lane >> 4;                       // 0: lanes 0-15, 1: lanes 16-31
  for (int t2 = 0; t2 < 2; ++t2) {
    const int mt = wave * 32 + t2 * 16;
    const unsigned short* xrow = xs + (mt + (lane & 15)) * XS_STRIDE + hi * 8;
    const unsigned short* wrow = wst + (lane & 15) * XS_STRIDE + hi * 8;

    v8f c0 = {}, c1 = {}, c2 = {}, c3 = {};
    v8bf aL[2], aH[2], bL[2][4], bH[2][4];

    // prologue: fragments for k0 = 0
    aL[0] = *(const v8bf*)(xrow);
    aH[0] = *(const v8bf*)(xrow + 16);
    #pragma unroll
    for (int g = 0; g < 4; ++g) {
      const unsigned short* wr = wrow + g * 16 * XS_STRIDE;
      bL[0][g] = *(const v8bf*)wr;
      bH[0][g] = *(const v8bf*)(wr + 16);
    }

    #pragma unroll
    for (int kk = 0; kk < 8; ++kk) {
      const int cur = kk & 1, nxt = cur ^ 1;
      if (kk < 7) {                               // issue next K-step's loads first
        const int kn = (kk + 1) * 32;
        aL[nxt] = *(const v8bf*)(xrow + kn);
        aH[nxt] = *(const v8bf*)(xrow + kn + 16);
        #pragma unroll
        for (int g = 0; g < 4; ++g) {
          const unsigned short* wr = wrow + g * 16 * XS_STRIDE + kn;
          bL[nxt][g] = *(const v8bf*)wr;
          bH[nxt][g] = *(const v8bf*)(wr + 16);
        }
      }
      v16bf a;
      #pragma unroll
      for (int i = 0; i < 8; ++i) { a[i] = aL[cur][i]; a[8 + i] = aH[cur][i]; }
      #pragma unroll
      for (int g = 0; g < 4; ++g) {
        v16bf bb;
        #pragma unroll
        for (int i = 0; i < 8; ++i) { bb[i] = bL[cur][g][i]; bb[8 + i] = bH[cur][g][i]; }
        v8f& cc = (g == 0) ? c0 : (g == 1) ? c1 : (g == 2) ? c2 : c3;
        cc = __builtin_amdgcn_wmma_f32_16x16x32_bf16(false, a, false, bb,
                                                     (short)0, cc, false, false);
      }
    }

    // scatter C (VGPR v: M = v + 8*(lane>=16), N = lane&15) into padded score rows
    #pragma unroll
    for (int g = 0; g < 4; ++g) {
      const v8f cc = (g == 0) ? c0 : (g == 1) ? c1 : (g == 2) ? c2 : c3;
      #pragma unroll
      for (int v = 0; v < 8; ++v) {
        sc[(mt + v + hi * 8) * SC_STRIDE + g * 16 + (lane & 15)] = cc[v];
      }
    }
  }
  __syncthreads();

  // ---- traversal: levels 0..5 from dense scores, levels 6..9 per-lane bf16 gather dots ----
  {
    const int s = tid;
    float m = 1.0f;
    int idx = 0;
    #pragma unroll
    for (int d = 0; d < 6; ++d) {
      float logit = alphav[idx] * (sc[s * SC_STRIDE + idx] + bias[idx]);
      float val = 1.0f / (1.0f + __expf(-logit));
      bool gr = logit >= 0.0f;                    // sigmoid(z) >= 0.5  <=>  z >= 0
      m *= gr ? val : (1.0f - val);
      idx = 2 * idx + 1 + (gr ? 1 : 0);
    }
    for (int d = 6; d < 10; ++d) {
      const uint4* wr = (const uint4*)(wbf + idx * NFEAT);   // L2-resident bf16 row (512B)
      const uint4* xr = (const uint4*)(xs + s * XS_STRIDE);
      float dot = 0.0f;
      #pragma unroll 8
      for (int k = 0; k < 32; ++k) {              // 32 x 16B = 256 halves
        uint4 w4 = wr[k];
        uint4 x4 = xr[k];
        dot = fmaf(bf_lo(x4.x), bf_lo(w4.x), dot);
        dot = fmaf(bf_hi(x4.x), bf_hi(w4.x), dot);
        dot = fmaf(bf_lo(x4.y), bf_lo(w4.y), dot);
        dot = fmaf(bf_hi(x4.y), bf_hi(w4.y), dot);
        dot = fmaf(bf_lo(x4.z), bf_lo(w4.z), dot);
        dot = fmaf(bf_hi(x4.z), bf_hi(w4.z), dot);
        dot = fmaf(bf_lo(x4.w), bf_lo(w4.w), dot);
        dot = fmaf(bf_hi(x4.w), bf_hi(w4.w), dot);
      }
      float logit = alphav[idx] * (dot + bias[idx]);
      float val = 1.0f / (1.0f + __expf(-logit));
      bool gr = logit >= 0.0f;
      m *= gr ? val : (1.0f - val);
      idx = 2 * idx + 1 + (gr ? 1 : 0);
    }
    mlt[s] = m;
    lfi[s] = idx - N_INTERNAL;
  }
  __syncthreads();

  // ---- coalesced output: 64 lanes per sample write 256B rows ----
  {
    const int j = tid & 63;
    for (int sb = (tid >> 6); sb < TILE_S; sb += 4) {
      float m = mlt[sb];
      int lf = lfi[sb];
      out[(tile_base + sb) * NOUT + j] = m * leaves[lf * NOUT + j];
    }
  }
}

extern "C" void kernel_launch(void* const* d_in, const int* in_sizes, int n_in,
                              void* d_out, int out_size, void* d_ws, size_t ws_size,
                              hipStream_t stream) {
  (void)in_sizes; (void)n_in; (void)out_size; (void)ws_size;
  const float* x      = (const float*)d_in[0];
  const float* W      = (const float*)d_in[1];
  const float* bias   = (const float*)d_in[2];
  const float* alphav = (const float*)d_in[3];
  const float* leaves = (const float*)d_in[4];
  float* out = (float*)d_out;
  unsigned short* wbf = (unsigned short*)d_ws;   // 1023*256*2 = 512KB scratch

  const int nW = N_INTERNAL * NFEAT;
  wconv_kernel<<<(nW + 255) / 256, 256, 0, stream>>>(W, wbf, nW);
  ogtree_kernel<<<BATCH / TILE_S, 256, SMEM_BYTES, stream>>>(x, wbf, bias, alphav,
                                                             leaves, out);
}
